// HierarchicalActionEncoder_31722628448613
// MI455X (gfx1250) — matrix operand order repaired and checked by reading, not compiled, not run on hardware
//
#include <hip/hip_runtime.h>

typedef _Float16 h8   __attribute__((ext_vector_type(8)));
typedef _Float16 v16h __attribute__((ext_vector_type(16)));
typedef float    v8f  __attribute__((ext_vector_type(8)));

#define N_ROWS   262144            // 4096*64
#define TILE_M   32
#define N_TILES  (N_ROWS / TILE_M) // 8192
#define NWG      512
#define NTHREADS 256

// Layer geometry (zero-padded block-diagonal formulation).
// KP = K (input, padded to x32), NP = N (output, padded to x16),
// OSTR = stride of output activation buffer = KP of the next layer.
constexpr int KP[7]   = { 64, 288, 160,  96, 128,  64,  32};
constexpr int NP[7]   = {288, 144,  80, 128,  64,  32, 256};
constexpr int OSTR[6] = {288, 160,  96, 128,  64,  32};
constexpr int WOFF[8] = {0, 18432, 59904, 72704, 84992, 93184, 95232, 103424};
constexpr int BOFF[8] = {0, 288, 432, 512, 640, 704, 736, 992};
#define W_TOTAL 103424
#define B_TOTAL 992

// Part-MLP block-diagonal placement tables (layers 0..2)
__device__ constexpr int P_FIN[3][6]   = {{7,7,9,9,12,12},{32,32,48,48,64,64},{16,16,24,24,32,32}};
__device__ constexpr int P_FOUT[3][6]  = {{32,32,48,48,64,64},{16,16,24,24,32,32},{8,8,12,12,16,16}};
__device__ constexpr int P_INOFF[3][6] = {{0,7,14,23,32,44},{0,32,64,112,160,224},{0,16,32,56,80,112}};
__device__ constexpr int P_OUTOFF[3][6]= {{0,32,64,112,160,224},{0,16,32,56,80,112},{0,8,16,28,40,56}};
__device__ constexpr int GFIN[4]  = {72,128,64,32};
__device__ constexpr int GFOUT[4] = {128,64,32,256};

struct Ptrs { const float* p[47]; };

// LDS byte layout
#define LDS_W    0        // 103424 halfs = 206848 B
#define LDS_B    206848   // 992 f32     =   3968 B
#define LDS_A0   210816   // 5120 halfs  =  10240 B (strides 64/160/128/32)
#define LDS_A1   221056   // 9216 halfs  =  18432 B (strides 288/96/64)
#define LDS_OB   239488   // 32x256 f32  =  32768 B
#define LDS_TOTAL 272256

// ---------------------------------------------------------------------------
// Prep: scatter reference params into f16 transposed block-diagonal Wt[n][k]
// (zero padded) + packed f32 biases, into d_ws. Deterministic, every launch.
// ---------------------------------------------------------------------------
__global__ __launch_bounds__(256) void ham_prep(Ptrs ptrs, _Float16* __restrict__ wOut,
                                                float* __restrict__ bOut) {
  const int idx = blockIdx.x * 256 + threadIdx.x;
  if (idx < W_TOTAL) {
    int L = 0;
    #pragma unroll
    for (int j = 0; j < 6; j++) if (idx >= WOFF[j + 1]) L = j + 1;
    const int local = idx - WOFF[L];
    const int K = KP[L];
    const int n = local / K, k = local - n * K;
    float v = 0.f;
    if (L < 3) {
      #pragma unroll
      for (int p = 0; p < 6; p++) {
        const int oo = P_OUTOFF[L][p], fo = P_FOUT[L][p];
        if (n >= oo && n < oo + fo) {
          const int io = P_INOFF[L][p], fi = P_FIN[L][p];
          if (k >= io && k < io + fi)
            v = ptrs.p[1 + p * 6 + L * 2][(k - io) * fo + (n - oo)];
        }
      }
    } else {
      const int gl = L - 3;
      if (k < GFIN[gl] && n < GFOUT[gl])
        v = ptrs.p[37 + gl * 2][k * GFOUT[gl] + n];
    }
    wOut[idx] = (_Float16)v;
  } else if (idx < W_TOTAL + B_TOTAL) {
    const int bi = idx - W_TOTAL;
    int L = 0;
    #pragma unroll
    for (int j = 0; j < 6; j++) if (bi >= BOFF[j + 1]) L = j + 1;
    const int n = bi - BOFF[L];
    float v = 0.f;
    if (L < 3) {
      #pragma unroll
      for (int p = 0; p < 6; p++) {
        const int oo = P_OUTOFF[L][p], fo = P_FOUT[L][p];
        if (n >= oo && n < oo + fo) v = ptrs.p[2 + p * 6 + L * 2][n - oo];
      }
    } else {
      v = ptrs.p[38 + (L - 3) * 2][n];
    }
    bOut[bi] = v;
  }
}

// ---------------------------------------------------------------------------
// Main fused kernel: persistent workgroups, WMMA f16 GEMM chain + LayerNorm.
// ---------------------------------------------------------------------------
__global__ __launch_bounds__(NTHREADS) void ham_main(const float* __restrict__ xin,
    const _Float16* __restrict__ wsW, const float* __restrict__ wsB,
    const float* __restrict__ gamma, const float* __restrict__ beta,
    float* __restrict__ out) {
  extern __shared__ char smem[];
  _Float16* wlds = (_Float16*)(smem + LDS_W);
  float*    blds = (float*)   (smem + LDS_B);
  _Float16* A0   = (_Float16*)(smem + LDS_A0);
  _Float16* A1   = (_Float16*)(smem + LDS_A1);
  float*    obuf = (float*)   (smem + LDS_OB);

  const int tid  = threadIdx.x;
  const int lane = tid & 31;
  // Force wave index scalar-uniform: keeps the per-wave tile loops as scalar
  // branches (no exec-mask loop) so EXEC is structurally all-1s at every WMMA.
  const int wave = __builtin_amdgcn_readfirstlane(tid >> 5);
  const int row  = lane & 15;   // M / N index within half-wave
  const int hi   = lane >> 4;   // which half of the wave

  // Stage all weights (16B vectors) + biases into LDS, once per workgroup.
  {
    const uint4* src = (const uint4*)wsW;
    uint4* dst = (uint4*)wlds;
    for (int i = tid; i < W_TOTAL / 8; i += NTHREADS) dst[i] = src[i];
    for (int i = tid; i < B_TOTAL; i += NTHREADS) blds[i] = wsB[i];
  }
  __syncthreads();

  for (int tile = blockIdx.x; tile < N_TILES; tile += NWG) {
    // Stage 32 rows x 56 f32 -> A0 (f16, stride 64, zero-padded cols 56..63)
    {
      const float* xr = xin + (size_t)tile * TILE_M * 56;
      for (int i = tid; i < TILE_M * 64; i += NTHREADS) {
        const int r = i >> 6, c = i & 63;
        A0[i] = (c < 56) ? (_Float16)xr[r * 56 + c] : (_Float16)0.f;
      }
      // Prefetch next tile's input (global_prefetch_b8) behind the compute.
      if (tile + NWG < N_TILES) {
        const char* nx = (const char*)(xin + (size_t)(tile + NWG) * TILE_M * 56);
        __builtin_prefetch(nx + tid * 28, 0, 1);
      }
    }
    __syncthreads();

    // 7 GEMM layers (fully unrolled so all geometry const-folds)
    #pragma unroll
    for (int L = 0; L < 7; L++) {
      const int K = KP[L], N = NP[L];
      const _Float16* ib  = (L & 1) ? A1 : A0;
      _Float16* ob16      = (L < 6) ? ((L & 1) ? A0 : A1) : nullptr;
      const int OS        = (L < 6) ? OSTR[L] : 256;
      const _Float16* Wl  = wlds + WOFF[L];
      const float*    Bl  = blds + BOFF[L];
      const int TN = N >> 4;

      for (int t = wave; t < 2 * TN; t += NTHREADS / 32) {
        const int mt = t / TN, nt = t - mt * TN;
        v8f acc = {};
        #pragma unroll
        for (int kk = 0; kk < K; kk += 32) {
          // A fragment (16x32 f16): lane<16 -> M=row, K = kk+[0..7],[16..23]
          //                         lane>=16 -> M=row, K = kk+[8..15],[24..31]
          const _Float16* ap = ib + (mt * 16 + row) * K + kk + hi * 8;
          h8 a0 = *(const h8*)(ap);
          h8 a1 = *(const h8*)(ap + 16);
          // B fragment (32x16 f16), weights stored transposed Wt[n][k]:
          // lane<16 -> N=row, K = kk+[0..7],[16..23]; lane>=16 -> K = kk+[8..15],[24..31]
          const _Float16* bp = Wl + (nt * 16 + row) * K + kk + hi * 8;
          h8 b0 = *(const h8*)(bp);
          h8 b1 = *(const h8*)(bp + 16);
          v16h av, bv;
          #pragma unroll
          for (int i = 0; i < 8; i++) {
            av[i] = a0[i]; av[i + 8] = a1[i];
            bv[i] = b0[i]; bv[i + 8] = b1[i];
          }
          acc = __builtin_amdgcn_wmma_f32_16x16x32_f16(
              false, av, false, bv, (short)0, acc, false, false);
        }
        // bias + activation + writeback (C layout: VGPR i -> M = i + hi*8)
        const float bias = Bl[nt * 16 + row];
        #pragma unroll
        for (int i = 0; i < 8; i++) {
          float x = acc[i] + bias;
          if (L != 2 && L != 6) x = fmaxf(x, 0.f);
          const int m = mt * 16 + i + hi * 8;
          if (L < 6) ob16[m * OS + nt * 16 + row] = (_Float16)x;
          else       obuf[m * 256 + nt * 16 + row] = x;
        }
      }
      // Zero the K-padding columns of the output buffer (layers 1 and 2)
      if (L < 6) {
        const int pad = OS - N;
        if (pad > 0) {
          for (int i = tid; i < TILE_M * pad; i += NTHREADS) {
            const int r = i / pad, c = i - r * pad;
            ob16[r * OS + N + c] = (_Float16)0.f;
          }
        }
      }
      __syncthreads();
    }

    // LayerNorm over 256, one wave handles 4 rows, coalesced global store.
    #pragma unroll 1
    for (int rr = 0; rr < 4; rr++) {
      const int r = wave * 4 + rr;
      float s = 0.f, s2 = 0.f, vals[8];
      #pragma unroll
      for (int j = 0; j < 8; j++) {
        const float v = obuf[r * 256 + j * 32 + lane];
        vals[j] = v; s += v; s2 += v * v;
      }
      #pragma unroll
      for (int m = 1; m < 32; m <<= 1) {
        s  += __shfl_xor(s,  m, 32);
        s2 += __shfl_xor(s2, m, 32);
      }
      const float mu  = s * (1.f / 256.f);
      const float var = s2 * (1.f / 256.f) - mu * mu;
      const float inv = rsqrtf(var + 1e-5f);
      const size_t base = ((size_t)tile * TILE_M + r) * 256;
      #pragma unroll
      for (int j = 0; j < 8; j++) {
        const int c = j * 32 + lane;
        out[base + c] = (vals[j] - mu) * inv * gamma[c] + beta[c];
      }
    }
    __syncthreads();
  }
}

extern "C" void kernel_launch(void* const* d_in, const int* in_sizes, int n_in,
                              void* d_out, int out_size, void* d_ws, size_t ws_size,
                              hipStream_t stream) {
  (void)in_sizes; (void)out_size; (void)ws_size;
  Ptrs P;
  for (int i = 0; i < 47; i++) P.p[i] = (i < n_in) ? (const float*)d_in[i] : nullptr;

  _Float16* wsW = (_Float16*)d_ws;
  float*    wsB = (float*)((char*)d_ws + 206848);

  const int prepBlocks = (W_TOTAL + B_TOTAL + 255) / 256;
  ham_prep<<<prepBlocks, 256, 0, stream>>>(P, wsW, wsB);

  (void)hipFuncSetAttribute((const void*)ham_main,
                            hipFuncAttributeMaxDynamicSharedMemorySize, LDS_TOTAL);
  ham_main<<<NWG, NTHREADS, LDS_TOTAL, stream>>>(
      (const float*)d_in[0], wsW, wsB,
      (const float*)d_in[45], (const float*)d_in[46], (float*)d_out);
}
